// CliqueGNN_66838281061251
// MI455X (gfx1250) — matrix operand order repaired and checked by reading, not compiled, not run on hardware
//
#include <hip/hip_runtime.h>

// CliqueGNN forward, restructured for MI455X (gfx1250):
//  - Per-edge linears algebraically decomposed: only one [E,128]x[128,128] GEMM
//    per layer at edge scale, executed with v_wmma_f32_16x16x32_f16 (fp32 acc).
//  - Edge tensors stored as f16 (37.7MB each) -> whole working set is L2-resident
//    (192MB L2); traffic ~0.25GB vs 23.3TB/s HBM.
//  - BN batch statistics over E rows fused into the WMMA epilogue
//    (register partials -> ds_add_f32 -> few global atomics).
//  - Gather/scatter aggregation done generally with LDS-privatized histograms.
//
// Input order assumption: setup_inputs() dict insertion order, tuples in order:
//  0 edge_index[2,E] i32, 1 edge_attr[E,3] f32,
//  2/3 node_emb w,b  4/5 edge_emb w,b  6/7 policy w,b  8/9 v1 w,b  10/11 v2 w,b
//  then per layer i (base=12+12*i): gcn w,b | gcn_bn g,b | eb_edge w,b |
//  eb_node w[256,128],b | eb_comb w[256,128],b | eb_bn g,b

typedef _Float16 half_t;
typedef __attribute__((ext_vector_type(16))) _Float16 v16h;
typedef __attribute__((ext_vector_type(8)))  float    v8f;

#define NVERT 384
#define H     128
#define EE    (NVERT*NVERT)          // 147456
#define PP    (NVERT*(NVERT-1)/2)    // 73536
#define BN_EPS 1e-5f

// ---------------------------------------------------------------- small utils
__global__ void zeroK(float* p, int n){
  int t = blockIdx.x*blockDim.x + threadIdx.x;
  if (t < n) p[t] = 0.f;
}

__global__ void nodeEmbK(float* x, const float* w, const float* b){
  int t = blockIdx.x*blockDim.x + threadIdx.x;
  if (t < NVERT*H){ int n = t >> 7, h = t & 127; x[t] = (float)n * w[h] + b[h]; }
}

// ef[e][h] = sum_k attr[e][k]*w[k][h] + b[h], stored f16, 8 cols per thread
__global__ void edgeEmbK(half_t* ef, const float* attr, const float* w, const float* b){
  int g = blockIdx.x*blockDim.x + threadIdx.x;   // EE*16 threads
  int e = g >> 4, c8 = (g & 15) * 8;
  float a0 = attr[e*3+0], a1 = attr[e*3+1], a2 = attr[e*3+2];
  union { uint4 u; half_t h[8]; } o;
#pragma unroll
  for (int j = 0; j < 8; j++){
    int hh = c8 + j;
    o.h[j] = (half_t)(a0*w[hh] + a1*w[H+hh] + a2*w[2*H+hh] + b[hh]);
  }
  ((uint4*)ef)[g] = o.u;
}

// degree histogram: LDS-privatized, then merge
__global__ void degK(float* deg, const int* dstI){
  __shared__ float sdeg[NVERT];
  for (int i = threadIdx.x; i < NVERT; i += blockDim.x) sdeg[i] = 0.f;
  __syncthreads();
  for (int e = blockIdx.x*blockDim.x + threadIdx.x; e < EE; e += gridDim.x*blockDim.x)
    atomicAdd(&sdeg[dstI[e]], 1.0f);
  __syncthreads();
  for (int i = threadIdx.x; i < NVERT; i += blockDim.x) atomicAdd(&deg[i], sdeg[i]);
}

__global__ void dinvK(float* dinv, const float* deg){
  int t = blockIdx.x*blockDim.x + threadIdx.x;
  if (t < NVERT){ float d = deg[t]; dinv[t] = d > 0.f ? rsqrtf(fmaxf(d, 1.f)) : 0.f; }
}

// generic small GEMM  C[M,128] = A[M,128] @ B[128,128]
__global__ void gemmK(float* C, const float* A, const float* B, int M){
  int t = blockIdx.x*blockDim.x + threadIdx.x;
  if (t >= M*H) return;
  int m = t >> 7, n = t & 127;
  const float* a = A + m*H;
  float s = 0.f;
#pragma unroll 8
  for (int k = 0; k < H; k++) s += a[k]*B[k*H+n];
  C[t] = s;
}

__global__ void initAggK(float* agg, const float* b){
  int t = blockIdx.x*blockDim.x + threadIdx.x;
  if (t < NVERT*H) agg[t] = b[t & 127];
}

// agg[dst] += xw[src]*dinv[src]*dinv[dst]  -- LDS-privatized in 96-node chunks
#define AGG_CH 96
__global__ void __launch_bounds__(256) aggScatterK(float* agg, const float* xw,
    const int* srcI, const int* dstI, const float* dinv){
  __shared__ float sAgg[AGG_CH*H];                       // 48KB
  int tid = threadIdx.x, lane = tid & 31;
  int wid = blockIdx.x*8 + (tid >> 5);
  int nw  = gridDim.x*8;
  for (int c = 0; c < 4; c++){
    for (int i = tid; i < AGG_CH*H; i += 256) sAgg[i] = 0.f;
    __syncthreads();
    int lo = c*AGG_CH;
    for (int e = wid; e < EE; e += nw){
      int d = dstI[e];                                   // uniform across wave
      int dl = d - lo;
      if ((unsigned)dl < AGG_CH){
        int s = srcI[e];
        float nrm = dinv[s]*dinv[d];
        float4 v = ((const float4*)(xw + (size_t)s*H))[lane];
        float* p = sAgg + dl*H + lane*4;
        atomicAdd(p+0, v.x*nrm); atomicAdd(p+1, v.y*nrm);
        atomicAdd(p+2, v.z*nrm); atomicAdd(p+3, v.w*nrm);
      }
    }
    __syncthreads();
    for (int i = tid; i < AGG_CH*H; i += 256) atomicAdd(&agg[lo*H + i], sAgg[i]);
    __syncthreads();
  }
}

// node BN (batch stats over 384 rows) + ReLU
__global__ void bnReluNodeK(float* xn, const float* agg, const float* g, const float* b){
  int h = threadIdx.x;
  float s = 0.f, q = 0.f;
  for (int n = 0; n < NVERT; n++){ float v = agg[n*H+h]; s += v; q += v*v; }
  float mu = s/NVERT, var = q/NVERT - mu*mu;
  float a = g[h]*rsqrtf(var + BN_EPS);
  float c = b[h] - mu*a;
  for (int n = 0; n < NVERT; n++){
    float v = agg[n*H+h]*a + c;
    xn[n*H+h] = v > 0.f ? v : 0.f;
  }
}

// bias_c[h] = bn@Wc_top + be@Wc_bot + bc
__global__ void biasCK(float* biasC, const float* bn_, const float* WcTop,
                       const float* be_, const float* WcBot, const float* bc_){
  int h = threadIdx.x;
  float s = bc_[h];
  for (int k = 0; k < H; k++) s += bn_[k]*WcTop[k*H+h];
  for (int k = 0; k < H; k++) s += be_[k]*WcBot[k*H+h];
  biasC[h] = s;
}

// Wmat = We @ Wc_bot, packed f16 into WMMA B-operand lane layout.
// 16-bit B 32x16: lanes 0-15 hold K {0..7,16..23} (col n), lanes 16-31: +8.
__global__ void packWK(half_t* wpack, const float* We, const float* Wc){
  int t = blockIdx.x*blockDim.x + threadIdx.x;          // 16384
  int k = t >> 7, n = t & 127;
  float s = 0.f;
#pragma unroll 8
  for (int j = 0; j < H; j++) s += We[k*H+j]*Wc[(H+j)*H+n];
  int kt = k >> 5, kk = k & 31, nt = n >> 4, nn = n & 15;
  int lane = (((kk >> 3) & 1) << 4) | nn;
  int jj   = (kk & 7) | ((kk >> 4) << 3);
  wpack[((kt*8 + nt)*32 + lane)*16 + jj] = (half_t)s;
}

// Big edge GEMM: comb[e] = ef[e]@Wmat + R[src]+S[dst]+bias_c, f16 out,
// fused BN column sum/sumsq.  Block = 8 waves x 16 rows = 128 edges.
__global__ void __launch_bounds__(256) edgeGemmK(const half_t* __restrict__ ef,
    const half_t* __restrict__ wpack, const float* __restrict__ Rn,
    const float* __restrict__ Sn, const float* __restrict__ biasC,
    const int* __restrict__ srcI, const int* __restrict__ dstI,
    half_t* __restrict__ comb, float* __restrict__ gSum, float* __restrict__ gSq)
{
  __shared__ __align__(32) half_t sB[16384];            // packed 128x128 weights
  __shared__ float sSum[H], sSq[H];
  int tid = threadIdx.x;
  {
    const uint4* s4 = (const uint4*)wpack;
    uint4* d4 = (uint4*)sB;
    for (int i = tid; i < 2048; i += 256) d4[i] = s4[i];
  }
  if (tid < H){ sSum[tid] = 0.f; sSq[tid] = 0.f; }
  __syncthreads();

  int lane = tid & 31, wave = tid >> 5;
  int laneHalf = lane >> 4, lane15 = lane & 15;
  int mBase = blockIdx.x*128 + wave*16;
  int rowA  = mBase + lane15;

  // A operand (16-bit 16x32): lanes 0-15: K {0..7,16..23}; lanes 16-31: +8
  v16h a[4];
  const uint4* efRow = (const uint4*)(ef + (size_t)rowA*H);
#pragma unroll
  for (int kt = 0; kt < 4; kt++){
    union { v16h v; uint4 u[2]; } ua;
    ua.u[0] = efRow[kt*4 + laneHalf];
    ua.u[1] = efRow[kt*4 + laneHalf + 2];
    a[kt] = ua.v;
  }
  // epilogue gathers: C layout row = mBase + v + 8*laneHalf
  int eIdx[8], sIdx[8], dIdx[8];
#pragma unroll
  for (int v = 0; v < 8; v++){
    int e = mBase + v + laneHalf*8;
    eIdx[v] = e; sIdx[v] = srcI[e]; dIdx[v] = dstI[e];
  }
  for (int nt = 0; nt < 8; nt++){
    v8f acc = {};
#pragma unroll
    for (int kt = 0; kt < 4; kt++){
      const v16h bop = *(const v16h*)(sB + ((kt*8 + nt)*32 + lane)*16);
      acc = __builtin_amdgcn_wmma_f32_16x16x32_f16(
          false, a[kt], false, bop, (short)0, acc, false, false);
    }
    int n = nt*16 + lane15;
    float bc = biasC[n];
    float ps = 0.f, pq = 0.f;
#pragma unroll
    for (int v = 0; v < 8; v++){
      float val = acc[v] + Rn[sIdx[v]*H + n] + Sn[dIdx[v]*H + n] + bc;
      comb[(size_t)eIdx[v]*H + n] = (half_t)val;
      ps += val; pq += val*val;
    }
    ps += __shfl_xor(ps, 16, 32);
    pq += __shfl_xor(pq, 16, 32);
    if (laneHalf == 0){ atomicAdd(&sSum[n], ps); atomicAdd(&sSq[n], pq); }
  }
  __syncthreads();
  if (tid < H){ atomicAdd(&gSum[tid], sSum[tid]); atomicAdd(&gSq[tid], sSq[tid]); }
}

__global__ void bnFinalizeK(float* a, float* c, const float* sum, const float* sq,
                            const float* g, const float* b){
  int h = threadIdx.x;
  float mu  = sum[h]/(float)EE;
  float var = sq[h]/(float)EE - mu*mu;
  float ah  = g[h]*rsqrtf(var + BN_EPS);
  a[h] = ah; c[h] = b[h] - mu*ah;
}

// ef = relu(comb*a + c), f16 in/out
__global__ void normReluK(half_t* ef, const half_t* comb, const float* a, const float* c){
  int g = blockIdx.x*blockDim.x + threadIdx.x;          // EE*16
  int c8 = (g & 15)*8;
  union { uint4 u; half_t h[8]; } in, o;
  in.u = ((const uint4*)comb)[g];
#pragma unroll
  for (int j = 0; j < 8; j++){
    float v = (float)in.h[j]*a[c8+j] + c[c8+j];
    o.h[j] = (half_t)(v > 0.f ? v : 0.f);
  }
  ((uint4*)ef)[g] = o.u;
}

// scores = ef@wp + bp, scattered to upper-tri logits (wave per edge)
__global__ void __launch_bounds__(256) policyK(float* logits, const half_t* ef,
    const float* wp, const float* bp, const int* srcI, const int* dstI){
  int lane = threadIdx.x & 31;
  int e = blockIdx.x*8 + (threadIdx.x >> 5);
  if (e >= EE) return;
  const half_t* p = ef + (size_t)e*H + lane*4;
  float s = (float)p[0]*wp[lane*4]   + (float)p[1]*wp[lane*4+1]
          + (float)p[2]*wp[lane*4+2] + (float)p[3]*wp[lane*4+3];
#pragma unroll
  for (int off = 16; off > 0; off >>= 1) s += __shfl_down(s, off, 32);
  if (lane == 0){
    int si = srcI[e], di = dstI[e];
    if (si < di){
      int tri = si*(2*NVERT - si - 1)/2 + (di - si - 1);
      logits[tri] = s + bp[0];
    }
  }
}

__global__ void softmaxK(float* out, const float* logits){
  __shared__ float red[1024];
  __shared__ float s_mx, s_sum;
  int t = threadIdx.x;
  float mx = -3.4e38f;
  for (int i = t; i < PP; i += 1024) mx = fmaxf(mx, logits[i]);
  red[t] = mx; __syncthreads();
  for (int o = 512; o > 0; o >>= 1){ if (t < o) red[t] = fmaxf(red[t], red[t+o]); __syncthreads(); }
  if (t == 0) s_mx = red[0];
  __syncthreads();
  float sum = 0.f;
  for (int i = t; i < PP; i += 1024) sum += expf(logits[i] - s_mx);
  red[t] = sum; __syncthreads();
  for (int o = 512; o > 0; o >>= 1){ if (t < o) red[t] += red[t+o]; __syncthreads(); }
  if (t == 0) s_sum = red[0];
  __syncthreads();
  float inv = 1.f/s_sum;
  for (int i = t; i < PP; i += 1024) out[i] = expf(logits[i] - s_mx)*inv;
}

__global__ void valueK(float* out, const float* x, const float* w1, const float* b1,
                       const float* w2, const float* b2){
  __shared__ float sm[H];
  __shared__ float sh[64];
  int t = threadIdx.x;
  float s = 0.f;
  for (int n = 0; n < NVERT; n++) s += x[n*H + t];
  sm[t] = s/(float)NVERT;
  __syncthreads();
  if (t < 64){
    float a = b1[t];
    for (int h = 0; h < H; h++) a += sm[h]*w1[h*64 + t];
    sh[t] = a > 0.f ? a : 0.f;
  }
  __syncthreads();
  if (t == 0){
    float a = b2[0];
    for (int j = 0; j < 64; j++) a += sh[j]*w2[j];
    out[0] = tanhf(a);
  }
}

// ---------------------------------------------------------------------- host
extern "C" void kernel_launch(void* const* d_in, const int* in_sizes, int n_in,
                              void* d_out, int out_size, void* d_ws, size_t ws_size,
                              hipStream_t stream)
{
  (void)in_sizes; (void)n_in; (void)out_size; (void)ws_size;
  const int*   ei    = (const int*)d_in[0];
  const int*   srcI  = ei;
  const int*   dstI  = ei + EE;
  const float* eAttr = (const float*)d_in[1];
  const float* nW = (const float*)d_in[2],  *nB = (const float*)d_in[3];
  const float* eW = (const float*)d_in[4],  *eB = (const float*)d_in[5];
  const float* pW = (const float*)d_in[6],  *pB = (const float*)d_in[7];
  const float* v1W= (const float*)d_in[8],  *v1B= (const float*)d_in[9];
  const float* v2W= (const float*)d_in[10], *v2B= (const float*)d_in[11];
  float* out = (float*)d_out;

  char* wsc = (char*)d_ws;
  auto carve = [&](size_t bytes) -> void* {
    void* p = (void*)wsc;
    wsc += (bytes + 255) & ~(size_t)255;
    return p;
  };
  half_t* ef    = (half_t*)carve((size_t)EE*H*2);
  half_t* comb  = (half_t*)carve((size_t)EE*H*2);
  float*  logits= (float*) carve((size_t)PP*4);
  float*  x0    = (float*) carve(NVERT*H*4);
  float*  x1    = (float*) carve(NVERT*H*4);
  float*  xw    = (float*) carve(NVERT*H*4);
  float*  agg   = (float*) carve(NVERT*H*4);
  float*  Pn    = (float*) carve(NVERT*H*4);
  float*  Qn    = (float*) carve(NVERT*H*4);
  float*  Rn    = (float*) carve(NVERT*H*4);
  float*  Sn    = (float*) carve(NVERT*H*4);
  half_t* wpack = (half_t*)carve(H*H*2);
  float*  biasC = (float*) carve(H*4);
  float*  gStats= (float*) carve(2*H*4);     // [0..127]=sum, [128..255]=sumsq
  float*  bnA   = (float*) carve(H*4);
  float*  bnC   = (float*) carve(H*4);
  float*  deg   = (float*) carve(NVERT*4);
  float*  dinv  = (float*) carve(NVERT*4);

  zeroK<<<(PP+255)/256, 256, 0, stream>>>(logits, PP);
  zeroK<<<2, 256, 0, stream>>>(deg, NVERT);
  nodeEmbK<<<192, 256, 0, stream>>>(x0, nW, nB);
  edgeEmbK<<<EE*16/256, 256, 0, stream>>>(ef, eAttr, eW, eB);
  degK<<<32, 256, 0, stream>>>(deg, dstI);
  dinvK<<<2, 256, 0, stream>>>(dinv, deg);

  float* xCur = x0; float* xNext = x1;
  for (int i = 0; i < 2; i++){
    int base = 12 + 12*i;
    const float* gcnW = (const float*)d_in[base+0];
    const float* gcnB = (const float*)d_in[base+1];
    const float* bnG  = (const float*)d_in[base+2];
    const float* bnB  = (const float*)d_in[base+3];
    const float* eeW  = (const float*)d_in[base+4];   // eb_edge [128,128]
    const float* eeB  = (const float*)d_in[base+5];
    const float* enW  = (const float*)d_in[base+6];   // eb_node [256,128]
    const float* enB  = (const float*)d_in[base+7];
    const float* ecW  = (const float*)d_in[base+8];   // eb_comb [256,128]
    const float* ecB  = (const float*)d_in[base+9];
    const float* ebG  = (const float*)d_in[base+10];
    const float* ebB  = (const float*)d_in[base+11];

    zeroK<<<1, 256, 0, stream>>>(gStats, 2*H);
    // GCN branch (uses old x)
    gemmK<<<192, 256, 0, stream>>>(xw, xCur, gcnW, NVERT);
    initAggK<<<192, 256, 0, stream>>>(agg, gcnB);
    aggScatterK<<<48, 256, 0, stream>>>(agg, xw, srcI, dstI, dinv);
    bnReluNodeK<<<1, 128, 0, stream>>>(xNext, agg, bnG, bnB);
    // EdgeBlock decomposition (uses old x)
    gemmK<<<192, 256, 0, stream>>>(Pn, xCur, enW,       NVERT);
    gemmK<<<192, 256, 0, stream>>>(Qn, xCur, enW + H*H, NVERT);
    gemmK<<<192, 256, 0, stream>>>(Rn, Pn, ecW, NVERT);
    gemmK<<<192, 256, 0, stream>>>(Sn, Qn, ecW, NVERT);
    biasCK<<<1, 128, 0, stream>>>(biasC, enB, ecW, eeB, ecW + H*H, ecB);
    packWK<<<64, 256, 0, stream>>>(wpack, eeW, ecW);
    edgeGemmK<<<EE/128, 256, 0, stream>>>(ef, wpack, Rn, Sn, biasC, srcI, dstI,
                                          comb, gStats, gStats + H);
    bnFinalizeK<<<1, 128, 0, stream>>>(bnA, bnC, gStats, gStats + H, ebG, ebB);
    normReluK<<<EE*16/256, 256, 0, stream>>>(ef, comb, bnA, bnC);
    float* t = xCur; xCur = xNext; xNext = t;
  }

  policyK<<<EE/8, 256, 0, stream>>>(logits, ef, pW, pB, srcI, dstI);
  softmaxK<<<1, 1024, 0, stream>>>(out, logits);
  valueK<<<1, 128, 0, stream>>>(out + PP, xCur, v1W, v1B, v2W, v2B);
}
